// Decoder_39273180954717
// MI455X (gfx1250) — matrix operand order repaired and checked by reading, not compiled, not run on hardware
//
#include <hip/hip_runtime.h>
#include <hip/hip_bf16.h>

// ---------------------------------------------------------------------------
// GRU decoder with encoder + word attention for gfx1250 (MI455X).
// All GEMM operands are staged as zero-padded f16 so the WMMA GEMM hot loop is
// branch-free: contiguous b128 fragment loads + v_wmma_f32_16x16x32_f16 with
// fp32 accumulation, register-blocked 4x2 tiles per wave (B fragments reused
// 4x along M -- key for the L2-bound recurrent h @ W_hh^T over 132 sequential
// steps). Producers (gates / attention / output GEMM) emit f16 activations
// directly so no conversion ever sits on the critical path.
// ---------------------------------------------------------------------------

typedef __attribute__((ext_vector_type(16))) _Float16 v16h;
typedef __attribute__((ext_vector_type(8)))  float    v8f;

// Problem constants (from the reference)
#define CS   128          // encoder sequence length
#define CB   64           // batch
#define CE   1024         // encoder dim
#define CH   1024         // hidden
#define CO   135          // pose dim
#define CWL  64           // word length
#define CPL  32           // previous poses
#define CT   100          // generated steps
#define CWF  200          // word feature dim
#define KIH  160          // O=135 padded to mult of 32 (gi GEMM K)
#define KWA  224          // 200 padded (word GEMM K)
#define NWO  144          // O=135 padded to mult of 16 (out GEMM N)

// ---------------------------------------------------------------------------
// f32 (N x K) -> f16 (Np x Kp) with zero padding
// ---------------------------------------------------------------------------
__global__ void cvt_pad_f16(const float* __restrict__ W, _Float16* __restrict__ Wh,
                            int N, int K, int Np, int Kp) {
    int idx = blockIdx.x * blockDim.x + threadIdx.x;
    if (idx >= Np * Kp) return;
    int n = idx / Kp, k = idx % Kp;
    Wh[idx] = (n < N && k < K) ? (_Float16)W[(size_t)n * K + k] : (_Float16)0.f;
}

__global__ void zero_f16(_Float16* __restrict__ p, int n) {
    int i = blockIdx.x * blockDim.x + threadIdx.x;
    if (i < n) p[i] = (_Float16)0.f;
}

// ---------------------------------------------------------------------------
// encoder_hidden (2,B,E/2) -> f16 (B, E): row b = concat(eh[0,b,:], eh[1,b,:])
// ---------------------------------------------------------------------------
__global__ void gather_eh(const float* __restrict__ eh, _Float16* __restrict__ out) {
    int idx = blockIdx.x * blockDim.x + threadIdx.x;   // b*E + k
    if (idx >= CB * CE) return;
    int b = idx >> 10, k = idx & (CE - 1);
    const int half = CE / 2;
    float v = (k < half) ? eh[(size_t)b * half + k]
                         : eh[(size_t)CB * half + (size_t)b * half + (k - half)];
    out[idx] = (_Float16)v;
}

// ---------------------------------------------------------------------------
// Register-blocked WMMA GEMM: C = A(f16, M x Kp, lda) @ W^T + bias.
// W f16 (Np x Kp) row-major; Kp mult of 32; Np mult of 16; M mult of 16*MB.
// Dual optional outputs: Cf (f32, ldc) and Ch (f16, ldch); n < N guarded.
// Fragment layouts (wave32, 16-bit):
//   A 16x32: lane l<16 -> row M=l, K 0..15; lane>=16 -> K 16..31
//   B 32x16: lane n<16 -> col N=n, K 0..15 ; lanes 16..31 -> K 16..31
//   C 16x16: VGPR r -> (M = r + 8*(lane>=16), N = lane%16)
// ---------------------------------------------------------------------------
template <int MB, int NB>
__global__ void gemm_wmma(const _Float16* __restrict__ A, int lda,
                          const _Float16* __restrict__ Bw, int Kp,
                          const float* __restrict__ bias,
                          float* __restrict__ Cf, int ldc,
                          _Float16* __restrict__ Ch, int ldch,
                          int M, int N, int tilesNB) {
    int wave = (int)((blockIdx.x * blockDim.x + threadIdx.x) >> 5);
    int lane = threadIdx.x & 31;
    int tmb = wave / tilesNB, tnb = wave % tilesNB;
    int m0 = tmb * (16 * MB), n0 = tnb * (16 * NB);
    if (m0 >= M) return;                       // wave-uniform exit
    int l16 = lane & 15;
    int hi  = lane >> 4;                       // 0 or 1

    const _Float16* Arow[MB];
    const _Float16* Brow[NB];
#pragma unroll
    for (int i = 0; i < MB; ++i) Arow[i] = A  + (size_t)(m0 + 16 * i + l16) * lda;
#pragma unroll
    for (int j = 0; j < NB; ++j) Brow[j] = Bw + (size_t)(n0 + 16 * j + l16) * Kp;

    v8f c[MB][NB] = {};
    for (int kb = 0; kb < Kp; kb += 32) {
        int ka0 = kb + hi * 16;                // A: 16 contiguous halves
        int kb0 = kb + hi * 16;                // B: 16 contiguous halves
        v16h a[MB], b[NB];
#pragma unroll
        for (int i = 0; i < MB; ++i) {
#pragma unroll
            for (int q = 0; q < 8; ++q) {
                a[i][q]     = Arow[i][ka0 + q];
                a[i][8 + q] = Arow[i][ka0 + 8 + q];
            }
        }
#pragma unroll
        for (int j = 0; j < NB; ++j) {
#pragma unroll
            for (int q = 0; q < 16; ++q) b[j][q] = Brow[j][kb0 + q];
        }
#pragma unroll
        for (int i = 0; i < MB; ++i)
#pragma unroll
            for (int j = 0; j < NB; ++j)
                c[i][j] = __builtin_amdgcn_wmma_f32_16x16x32_f16(
                    false, a[i], false, b[j], (short)0, c[i][j], false, false);
    }

#pragma unroll
    for (int j = 0; j < NB; ++j) {
        int n = n0 + 16 * j + l16;
        if (n < N) {
            float bv = bias ? bias[n] : 0.f;
#pragma unroll
            for (int i = 0; i < MB; ++i) {
#pragma unroll
                for (int r = 0; r < 8; ++r) {
                    int m = m0 + 16 * i + r + 8 * hi;
                    float v = c[i][j][r] + bv;
                    if (Cf) Cf[(size_t)m * ldc  + n] = v;
                    if (Ch) Ch[(size_t)m * ldch + n] = (_Float16)v;
                }
            }
        }
    }
}

// ---------------------------------------------------------------------------
// GRU gates: h' = (1-z)*n + z*h. Writes h (f32, in place), h_h (f16), and
// optionally the first H columns of cat_h (f16, B x 3H).
// ---------------------------------------------------------------------------
__global__ void gru_gate(const float* __restrict__ gi, const float* __restrict__ gh,
                         float* __restrict__ h, _Float16* __restrict__ h_h,
                         _Float16* __restrict__ cat_h) {
    int idx = blockIdx.x * blockDim.x + threadIdx.x;    // b*H + j
    if (idx >= CB * CH) return;
    int b = idx / CH, j = idx - b * CH;
    const float* gib = gi + (size_t)b * 3 * CH;
    const float* ghb = gh + (size_t)b * 3 * CH;
    float r  = 1.f / (1.f + __expf(-(gib[j]          + ghb[j])));
    float z  = 1.f / (1.f + __expf(-(gib[CH + j]     + ghb[CH + j])));
    float nn = tanhf(gib[2 * CH + j] + r * ghb[2 * CH + j]);
    float out = (1.f - z) * nn + z * h[idx];
    h[idx]   = out;
    h_h[idx] = (_Float16)out;
    if (cat_h) cat_h[(size_t)b * 3 * CH + j] = (_Float16)out;
}

// ---------------------------------------------------------------------------
// Fused attention for one batch element per block (256 thr = 8 waves):
//   scores[s] = dot(proj[s,b,:], h[b,:]);  w = softmax(scores);
//   cat_h[b, coloff + j] = sum_s proj[s,b,j] * w[s]      (proj/h/cat in f16)
// ---------------------------------------------------------------------------
__global__ void attend_fused(const _Float16* __restrict__ proj, int S,
                             const _Float16* __restrict__ h,
                             _Float16* __restrict__ cat_h, int coloff) {
    __shared__ float sw[CS];                    // S <= 128
    int b = blockIdx.x;
    int lane = threadIdx.x & 31, wv = threadIdx.x >> 5;
    const _Float16* hb = h + (size_t)b * CH;

    for (int s = wv; s < S; s += 8) {
        const _Float16* p = proj + ((size_t)s * CB + b) * CH;
        float acc = 0.f;
        for (int j = lane; j < CH; j += 32) acc += (float)p[j] * (float)hb[j];
#pragma unroll
        for (int off = 16; off > 0; off >>= 1) acc += __shfl_xor(acc, off, 32);
        if (lane == 0) sw[s] = acc;
    }
    __syncthreads();

    if (wv == 0) {
        float m = -1e30f;
        for (int s = lane; s < S; s += 32) m = fmaxf(m, sw[s]);
#pragma unroll
        for (int off = 16; off > 0; off >>= 1) m = fmaxf(m, __shfl_xor(m, off, 32));
        float sum = 0.f;
        for (int s = lane; s < S; s += 32) { float e = __expf(sw[s] - m); sw[s] = e; sum += e; }
#pragma unroll
        for (int off = 16; off > 0; off >>= 1) sum += __shfl_xor(sum, off, 32);
        float inv = 1.f / sum;
        for (int s = lane; s < S; s += 32) sw[s] *= inv;
    }
    __syncthreads();

    for (int j = threadIdx.x; j < CH; j += blockDim.x) {
        float acc = 0.f;
        for (int s = 0; s < S; ++s) acc += (float)proj[((size_t)s * CB + b) * CH + j] * sw[s];
        cat_h[(size_t)b * 3 * CH + coloff + j] = (_Float16)acc;
    }
}

// ---------------------------------------------------------------------------
// Host side
// ---------------------------------------------------------------------------
extern "C" void kernel_launch(void* const* d_in, const int* in_sizes, int n_in,
                              void* d_out, int out_size, void* d_ws, size_t ws_size,
                              hipStream_t stream) {
    (void)in_sizes; (void)n_in; (void)out_size; (void)ws_size;
    const float* enc_states = (const float*)d_in[0];   // (S,B,E)
    const float* enc_hidden = (const float*)d_in[1];   // (2,B,E/2)
    const float* prev_poses = (const float*)d_in[2];   // (PL,B,O)
    const float* words      = (const float*)d_in[3];   // (WL,B,200)
    // d_in[4] = real_poses_len (compile-time CT=100)
    const float* W_ed  = (const float*)d_in[5];
    const float* b_ed  = (const float*)d_in[6];
    const float* W_att = (const float*)d_in[7];
    const float* b_att = (const float*)d_in[8];
    const float* W_watt= (const float*)d_in[9];
    const float* b_watt= (const float*)d_in[10];
    const float* W_ih  = (const float*)d_in[11];
    const float* W_hh  = (const float*)d_in[12];
    const float* b_ih  = (const float*)d_in[13];
    const float* b_hh  = (const float*)d_in[14];
    const float* W_out = (const float*)d_in[15];
    const float* b_out = (const float*)d_in[16];
    float* out = (float*)d_out;

    // ---- carve workspace ----
    char* ws = (char*)d_ws;
    size_t off = 0;
    auto carve = [&](size_t bytes) -> void* {
        void* p = ws + off;
        off = (off + bytes + 255) & ~(size_t)255;
        return p;
    };
    // f16 weights
    _Float16* Wed_h  = (_Float16*)carve((size_t)CH * CE * 2);
    _Float16* Watt_h = (_Float16*)carve((size_t)CH * CE * 2);
    _Float16* Wwat_h = (_Float16*)carve((size_t)CH * KWA * 2);
    _Float16* Wih_h  = (_Float16*)carve((size_t)3 * CH * KIH * 2);
    _Float16* Whh_h  = (_Float16*)carve((size_t)3 * CH * CH * 2);
    _Float16* Wout_h = (_Float16*)carve((size_t)NWO * 3 * CH * 2);
    // f16 staged activations
    _Float16* encst_h = (_Float16*)carve((size_t)CS * CB * CE * 2);
    _Float16* words_h = (_Float16*)carve((size_t)CWL * CB * KWA * 2);
    _Float16* xprev_h = (_Float16*)carve((size_t)CPL * CB * KIH * 2);
    _Float16* xstep_h = (_Float16*)carve((size_t)CB * KIH * 2);
    _Float16* ehcat_h = (_Float16*)carve((size_t)CB * CE * 2);
    _Float16* h_h     = (_Float16*)carve((size_t)CB * CH * 2);
    _Float16* cat_h   = (_Float16*)carve((size_t)CB * 3 * CH * 2);
    _Float16* eproj_h = (_Float16*)carve((size_t)CS * CB * CH * 2);
    _Float16* wproj_h = (_Float16*)carve((size_t)CWL * CB * CH * 2);
    // f32 buffers
    float* hbuf = (float*)carve((size_t)CB * CH * 4);
    float* gi   = (float*)carve((size_t)CB * 3 * CH * 4);
    float* gh   = (float*)carve((size_t)CB * 3 * CH * 4);

    auto cvt = [&](const float* W, _Float16* Wh, int N, int K, int Np, int Kp) {
        int total = Np * Kp;
        cvt_pad_f16<<<(total + 255) / 256, 256, 0, stream>>>(W, Wh, N, K, Np, Kp);
    };
    auto gemm = [&](const _Float16* A, int lda, const _Float16* Bw, int Kp,
                    const float* bias, float* Cf, int ldc,
                    _Float16* Ch, int ldch, int M, int N) {
        int tilesN  = (N + 15) / 16;
        int tilesM  = (M + 15) / 16;
        int MB = (tilesM % 4 == 0) ? 4 : 1;
        int NB = (tilesN % 2 == 0) ? 2 : 1;
        int tilesNB = (tilesN + NB - 1) / NB;
        int waves   = ((tilesM + MB - 1) / MB) * tilesNB;
        int blocks  = (waves + 7) / 8;          // 8 waves (256 thr) per block
        if      (MB == 4 && NB == 2) gemm_wmma<4,2><<<blocks,256,0,stream>>>(A,lda,Bw,Kp,bias,Cf,ldc,Ch,ldch,M,N,tilesNB);
        else if (MB == 4 && NB == 1) gemm_wmma<4,1><<<blocks,256,0,stream>>>(A,lda,Bw,Kp,bias,Cf,ldc,Ch,ldch,M,N,tilesNB);
        else if (MB == 1 && NB == 2) gemm_wmma<1,2><<<blocks,256,0,stream>>>(A,lda,Bw,Kp,bias,Cf,ldc,Ch,ldch,M,N,tilesNB);
        else                         gemm_wmma<1,1><<<blocks,256,0,stream>>>(A,lda,Bw,Kp,bias,Cf,ldc,Ch,ldch,M,N,tilesNB);
    };

    // ---- one-time staging: weights + large activations -> padded f16 ----
    cvt(W_ed,  Wed_h,  CH,      CE,     CH,      CE);
    cvt(W_att, Watt_h, CH,      CE,     CH,      CE);
    cvt(W_watt,Wwat_h, CH,      CWF,    CH,      KWA);
    cvt(W_ih,  Wih_h,  3 * CH,  CO,     3 * CH,  KIH);
    cvt(W_hh,  Whh_h,  3 * CH,  CH,     3 * CH,  CH);
    cvt(W_out, Wout_h, CO,      3 * CH, NWO,     3 * CH);
    cvt(enc_states, encst_h, CS * CB,  CE,  CS * CB,  CE);
    cvt(words,      words_h, CWL * CB, CWF, CWL * CB, KWA);
    cvt(prev_poses, xprev_h, CPL * CB, CO,  CPL * CB, KIH);
    zero_f16<<<(CB * KIH + 255) / 256, 256, 0, stream>>>(xstep_h, CB * KIH);

    // ---- initial hidden: h = eh_cat @ W_ed^T + b_ed (f32 + f16 outputs) ----
    gather_eh<<<(CB * CE + 255) / 256, 256, 0, stream>>>(enc_hidden, ehcat_h);
    gemm(ehcat_h, CE, Wed_h, CE, b_ed, hbuf, CH, h_h, CH, CB, CH);

    // ---- one-time projections (f16 outputs only) ----
    gemm(encst_h, CE,  Watt_h, CE,  b_att,  nullptr, 0, eproj_h, CH, CS * CB,  CH);
    gemm(words_h, KWA, Wwat_h, KWA, b_watt, nullptr, 0, wproj_h, CH, CWL * CB, CH);

    // ---- warm-up: 32 GRU steps over previous_poses ----
    for (int t = 0; t < CPL; ++t) {
        const _Float16* x = xprev_h + (size_t)t * CB * KIH;
        gemm(x,   KIH, Wih_h, KIH, b_ih, gi, 3 * CH, nullptr, 0, CB, 3 * CH);
        gemm(h_h, CH,  Whh_h, CH,  b_hh, gh, 3 * CH, nullptr, 0, CB, 3 * CH);
        gru_gate<<<(CB * CH + 255) / 256, 256, 0, stream>>>(gi, gh, hbuf, h_h, nullptr);
    }

    // ---- generation: 100 steps ----
    for (int t = 0; t < CT; ++t) {
        const _Float16* x = (t == 0) ? xprev_h + (size_t)(CPL - 1) * CB * KIH
                                     : xstep_h;
        gemm(x,   KIH, Wih_h, KIH, b_ih, gi, 3 * CH, nullptr, 0, CB, 3 * CH);
        gemm(h_h, CH,  Whh_h, CH,  b_hh, gh, 3 * CH, nullptr, 0, CB, 3 * CH);
        gru_gate<<<(CB * CH + 255) / 256, 256, 0, stream>>>(gi, gh, hbuf, h_h, cat_h);
        attend_fused<<<CB, 256, 0, stream>>>(eproj_h, CS,  h_h, cat_h, CH);
        attend_fused<<<CB, 256, 0, stream>>>(wproj_h, CWL, h_h, cat_h, 2 * CH);
        // pose (f32 to d_out, f16 padded to xstep_h for the next step)
        gemm(cat_h, 3 * CH, Wout_h, 3 * CH, b_out,
             out + (size_t)t * CB * CO, CO, xstep_h, KIH, CB, CO);
    }
}
